// Decoder_52965536694668
// MI455X (gfx1250) — compile-verified, matrix operand order checked by reading, not run on hardware
//
#include <hip/hip_runtime.h>

#define T_LEN 1024
#define B_SZ  256
#define H_SZ  512
#define O_SZ  64

typedef __bf16 bf16;
typedef __attribute__((ext_vector_type(16))) __bf16 v16bf;
typedef __attribute__((ext_vector_type(8)))  __bf16 v8bf;
typedef __attribute__((ext_vector_type(8)))  float  v8f;

// ---------------------------------------------------------------------------
// Fragment loaders (CDNA5 wave32 WMMA layouts, ISA 7.12.2)
// A: 16x32 bf16 tile of a row-major [rows x ld] matrix.
//   lanes 0-15 : row = row0+lane,   elems 0..7 = K[k0..k0+7],  8..15 = K[k0+16..k0+23]
//   lanes 16-31: row = row0+lane-16, elems 0..7 = K[k0+8..+15], 8..15 = K[k0+24..+31]
// B (32x16, element (k,n) = W[n][k] of row-major N x K weight) has the mirrored
// layout with lane -> n, so the same loader serves both.
// ---------------------------------------------------------------------------
__device__ __forceinline__ v16bf load_frag16(const bf16* __restrict__ base,
                                             int ld, int row0, int k0) {
  const int lane = threadIdx.x & 31;
  const int hsel = lane >> 4;       // 0: lanes 0-15, 1: lanes 16-31
  const int r    = lane & 15;
  const bf16* p = base + (size_t)(row0 + r) * (size_t)ld + (size_t)(k0 + hsel * 8);
  v8bf lo = *(const v8bf*)(p);        // K[k0 + hsel*8 .. +7]
  v8bf hi = *(const v8bf*)(p + 16);   // K[k0 + 16 + hsel*8 .. +7]
  v16bf out;
#pragma unroll
  for (int i = 0; i < 8; ++i) { out[i] = lo[i]; out[i + 8] = hi[i]; }
  return out;
}

__device__ __forceinline__ v8f wmma_bf16(v16bf a, v16bf b, v8f c) {
  // D = A x B + C, fp32 accumulate
  return __builtin_amdgcn_wmma_f32_16x16x32_bf16(false, a, false, b,
                                                 (short)0, c, false, false);
}

__device__ __forceinline__ float sigmoidf_fast(float x) {
  return 1.0f / (1.0f + __expf(-x));
}

// ---------------------------------------------------------------------------
// Prep kernels
// ---------------------------------------------------------------------------
__global__ void cvt_f32_to_bf16(const float* __restrict__ in,
                                bf16* __restrict__ out, int n) {
  int i = blockIdx.x * blockDim.x + threadIdx.x;
  if (i < n) out[i] = (bf16)in[i];
}

__global__ void init_h_kernel(const float* __restrict__ cv,
                              float* __restrict__ hf, bf16* __restrict__ hb, int n) {
  int i = blockIdx.x * blockDim.x + threadIdx.x;
  if (i < n) { float v = cv[i]; hf[i] = v; hb[i] = (bf16)v; }
}

// ---------------------------------------------------------------------------
// One GRU time step. Grid: (B/16, H/16), block: 1 wave (32 threads).
// Block (bx, by) produces h_new tile rows [16bx,16bx+16) x cols [16by,16by+16).
// Blocks with by < O/16 additionally emit y_{t-1} from the fully-formed h_t.
// ---------------------------------------------------------------------------
__global__ __launch_bounds__(32) void gru_step_kernel(
    const bf16*  __restrict__ x_bf,      // [B,O] bf16 input (nullptr at t==0 -> x=0)
    const float* __restrict__ h_in,      // [B,H] fp32  (h_t)
    const bf16*  __restrict__ h_in_bf,   // [B,H] bf16
    float*       __restrict__ h_out,     // [B,H] fp32  (h_{t+1})
    bf16*        __restrict__ h_out_bf,  // [B,H] bf16
    const bf16*  __restrict__ wih_bf,    // [3H,O]
    const bf16*  __restrict__ whh_bf,    // [3H,H]
    const bf16*  __restrict__ wout_bf,   // [O,H]
    const float* __restrict__ b_ih,      // [3H]
    const float* __restrict__ b_hh,      // [3H]
    const float* __restrict__ b_out,     // [O]
    float*       __restrict__ y_prev)    // [B,O] = outputs[t-1] (nullptr at t==0)
{
  const int m0   = blockIdx.x * 16;
  const int n0   = blockIdx.y * 16;
  const int lane = threadIdx.x & 31;
  const int hsel = lane >> 4;
  const int r    = lane & 15;

  v8f acc_r = {}, acc_z = {}, acc_in = {}, acc_hn = {};

  // gi = x @ w_ih.T contribution (K = O = 64); skipped when x == 0 (t == 0)
  if (x_bf != nullptr) {
#pragma unroll
    for (int kt = 0; kt < O_SZ / 32; ++kt) {
      const int k0 = kt * 32;
      v16bf a = load_frag16(x_bf, O_SZ, m0, k0);
      acc_r  = wmma_bf16(a, load_frag16(wih_bf, O_SZ, n0,            k0), acc_r);
      acc_z  = wmma_bf16(a, load_frag16(wih_bf, O_SZ, n0 + H_SZ,     k0), acc_z);
      acc_in = wmma_bf16(a, load_frag16(wih_bf, O_SZ, n0 + 2 * H_SZ, k0), acc_in);
    }
  }

  // gh = h @ w_hh.T contribution (K = H = 512)
#pragma unroll 4
  for (int kt = 0; kt < H_SZ / 32; ++kt) {
    const int k0 = kt * 32;
    v16bf a = load_frag16(h_in_bf, H_SZ, m0, k0);
    acc_r  = wmma_bf16(a, load_frag16(whh_bf, H_SZ, n0,            k0), acc_r);
    acc_z  = wmma_bf16(a, load_frag16(whh_bf, H_SZ, n0 + H_SZ,     k0), acc_z);
    acc_hn = wmma_bf16(a, load_frag16(whh_bf, H_SZ, n0 + 2 * H_SZ, k0), acc_hn);
  }

  // Gating epilogue (fp32). D layout: vreg j -> row m0 + hsel*8 + j, col n0 + r.
  const int   col = n0 + r;
  const float brz = b_ih[col] + b_hh[col];
  const float bzz = b_ih[col + H_SZ] + b_hh[col + H_SZ];
  const float bin = b_ih[col + 2 * H_SZ];
  const float bhn = b_hh[col + 2 * H_SZ];

#pragma unroll
  for (int j = 0; j < 8; ++j) {
    const int row = m0 + hsel * 8 + j;
    const float rg = sigmoidf_fast(acc_r[j] + brz);
    const float zg = sigmoidf_fast(acc_z[j] + bzz);
    const float ng = tanhf(acc_in[j] + bin + rg * (acc_hn[j] + bhn));
    const float hp = h_in[(size_t)row * H_SZ + col];
    const float hv = (1.0f - zg) * ng + zg * hp;
    h_out[(size_t)row * H_SZ + col]    = hv;
    h_out_bf[(size_t)row * H_SZ + col] = (bf16)hv;
  }

  // y_{t-1} = h_t @ w_out.T + b_out (needs full rows of h_t, which is our input)
  if (y_prev != nullptr && blockIdx.y < O_SZ / 16) {
    const int o0 = blockIdx.y * 16;
    v8f acc = {};
#pragma unroll 4
    for (int kt = 0; kt < H_SZ / 32; ++kt) {
      const int k0 = kt * 32;
      v16bf a = load_frag16(h_in_bf, H_SZ, m0, k0);
      acc = wmma_bf16(a, load_frag16(wout_bf, H_SZ, o0, k0), acc);
    }
    const int   oc = o0 + r;
    const float bo = b_out[oc];
#pragma unroll
    for (int j = 0; j < 8; ++j) {
      const int row = m0 + hsel * 8 + j;
      y_prev[(size_t)row * O_SZ + oc] = acc[j] + bo;
    }
  }
}

// Final output projection: y_{T-1} = h_T @ w_out.T + b_out.
__global__ __launch_bounds__(32) void gru_out_kernel(
    const bf16*  __restrict__ h_bf,     // [B,H]
    const bf16*  __restrict__ wout_bf,  // [O,H]
    const float* __restrict__ b_out,    // [O]
    float*       __restrict__ y)        // [B,O]
{
  const int m0   = blockIdx.x * 16;
  const int o0   = blockIdx.y * 16;
  const int lane = threadIdx.x & 31;
  const int hsel = lane >> 4;
  const int r    = lane & 15;

  v8f acc = {};
#pragma unroll 4
  for (int kt = 0; kt < H_SZ / 32; ++kt) {
    const int k0 = kt * 32;
    v16bf a = load_frag16(h_bf, H_SZ, m0, k0);
    acc = wmma_bf16(a, load_frag16(wout_bf, H_SZ, o0, k0), acc);
  }
  const int   oc = o0 + r;
  const float bo = b_out[oc];
#pragma unroll
  for (int j = 0; j < 8; ++j) {
    const int row = m0 + hsel * 8 + j;
    y[(size_t)row * O_SZ + oc] = acc[j] + bo;
  }
}

// ---------------------------------------------------------------------------
// Host launch
// ---------------------------------------------------------------------------
extern "C" void kernel_launch(void* const* d_in, const int* in_sizes, int n_in,
                              void* d_out, int out_size, void* d_ws, size_t ws_size,
                              hipStream_t stream) {
  (void)in_sizes; (void)n_in; (void)out_size; (void)ws_size;

  const float* code_vec = (const float*)d_in[0];   // [1,B,H]
  const float* target   = (const float*)d_in[1];   // [T,B,O]
  const float* w_ih     = (const float*)d_in[2];   // [3H,O]
  const float* w_hh     = (const float*)d_in[3];   // [3H,H]
  const float* b_ih     = (const float*)d_in[4];   // [3H]
  const float* b_hh     = (const float*)d_in[5];   // [3H]
  const float* w_out    = (const float*)d_in[6];   // [O,H]
  const float* b_out    = (const float*)d_in[7];   // [O]
  float* out = (float*)d_out;                      // [T*B*O] outputs ++ [B*H] h_last

  const size_t BH  = (size_t)B_SZ * H_SZ;          // 131072
  const size_t BO  = (size_t)B_SZ * O_SZ;          // 16384
  const size_t TBO = (size_t)T_LEN * BO;           // 16777216

  // Workspace layout (~37 MB): h fp32 x2 | h bf16 x2 | wih | whh | wout | target bf16
  char* ws = (char*)d_ws;
  float* hf[2];
  hf[0] = (float*)ws;
  hf[1] = hf[0] + BH;
  bf16* hb[2];
  hb[0] = (bf16*)(hf[1] + BH);
  hb[1] = hb[0] + BH;
  bf16* wih_bf  = hb[1] + BH;
  bf16* whh_bf  = wih_bf + (size_t)3 * H_SZ * O_SZ;
  bf16* wout_bf = whh_bf + (size_t)3 * H_SZ * H_SZ;
  bf16* tgt_bf  = wout_bf + (size_t)O_SZ * H_SZ;

  // One-time (per call) conversions — all independent, issued in order on one stream.
  {
    const int n1 = 3 * H_SZ * O_SZ;
    cvt_f32_to_bf16<<<(n1 + 255) / 256, 256, 0, stream>>>(w_ih, wih_bf, n1);
    const int n2 = 3 * H_SZ * H_SZ;
    cvt_f32_to_bf16<<<(n2 + 255) / 256, 256, 0, stream>>>(w_hh, whh_bf, n2);
    const int n3 = O_SZ * H_SZ;
    cvt_f32_to_bf16<<<(n3 + 255) / 256, 256, 0, stream>>>(w_out, wout_bf, n3);
    const int n4 = (int)TBO;
    cvt_f32_to_bf16<<<(n4 + 255) / 256, 256, 0, stream>>>(target, tgt_bf, n4);
    init_h_kernel<<<((int)BH + 255) / 256, 256, 0, stream>>>(code_vec, hf[0], hb[0], (int)BH);
  }

  // Sequential recurrence: 1024 step kernels (graph-capturable stream launches).
  const dim3 grid(B_SZ / 16, H_SZ / 16);
  for (int t = 0; t < T_LEN; ++t) {
    const bf16* x  = (t == 0) ? nullptr : (tgt_bf + (size_t)(t - 1) * BO);
    float*      yp = (t == 0) ? nullptr : (out + (size_t)(t - 1) * BO);
    const int cur = t & 1;
    const int nxt = cur ^ 1;
    gru_step_kernel<<<grid, 32, 0, stream>>>(
        x, hf[cur], hb[cur], hf[nxt], hb[nxt],
        wih_bf, whh_bf, wout_bf, b_ih, b_hh, b_out, yp);
  }

  // Tail: y_{T-1} from h_T (lives in buffer 0 since T is even), then h_last.
  gru_out_kernel<<<dim3(B_SZ / 16, O_SZ / 16), 32, 0, stream>>>(
      hb[0], wout_bf, b_out, out + (TBO - BO));
  hipMemcpyAsync(out + TBO, hf[0], BH * sizeof(float),
                 hipMemcpyDeviceToDevice, stream);
}
